// SlicedFP8FusedSDPA_82076825026614
// MI455X (gfx1250) — matrix operand order repaired and checked by reading, not compiled, not run on hardware
//
#include <hip/hip_runtime.h>

// ---------------------------------------------------------------------------
// Flash-style causal SDPA for MI455X (gfx1250), BF16 WMMA pipeline.
// B=1, H=16, QL=2048, KVL=4096, D=128, prefix = KVL-QL = 2048.
// - v_wmma_f32_16x16x32_bf16 for both gemms (A/B fragments in contiguous k = 16*half + s order)
// - DPP8 / row_ror:8 butterflies for softmax reductions (no LDS traffic)
// - exp2-domain online softmax (v_exp_f32 directly)
// - register-resident prefetch of next KV tile overlapped with compute
// ---------------------------------------------------------------------------

typedef __bf16 v16bf __attribute__((ext_vector_type(16)));
typedef __bf16 v8bf  __attribute__((ext_vector_type(8)));
typedef __bf16 v4bf  __attribute__((ext_vector_type(4)));
typedef float  v8f   __attribute__((ext_vector_type(8)));

#define NHEADS 16
#define QLEN   2048
#define KVLEN  4096
#define DHEAD  128
#define PREFIX (KVLEN - QLEN)
#define BR     128            // q rows per workgroup (8 waves x 16 rows)
#define BC     64             // kv tile
#define NEGBIG (-1e30f)

static __device__ __forceinline__ __bf16 f2bf(float f) { return (__bf16)f; }

// xor-butterfly within 8-lane groups via DPP8 (pure VALU, no LDS traffic)
template <int M>
static __device__ __forceinline__ float dpp8_xor(float x) {
  constexpr int sel = ((0 ^ M) << 0)  | ((1 ^ M) << 3)  | ((2 ^ M) << 6) |
                      ((3 ^ M) << 9)  | ((4 ^ M) << 12) | ((5 ^ M) << 15) |
                      ((6 ^ M) << 18) | ((7 ^ M) << 21);
  return __builtin_bit_cast(float,
      __builtin_amdgcn_mov_dpp8(__builtin_bit_cast(int, x), sel));
}
// xor-8 within a 16-lane row == row_ror:8 (DPP16 ctrl 0x128)
static __device__ __forceinline__ float dpp16_xor8(float x) {
  return __builtin_bit_cast(float,
      __builtin_amdgcn_update_dpp(__builtin_bit_cast(int, x),
                                  __builtin_bit_cast(int, x),
                                  0x128, 0xF, 0xF, true));
}
static __device__ __forceinline__ float red_max16(float x) {
  x = fmaxf(x, dpp8_xor<1>(x));
  x = fmaxf(x, dpp8_xor<2>(x));
  x = fmaxf(x, dpp8_xor<4>(x));
  x = fmaxf(x, dpp16_xor8(x));
  return x;
}
static __device__ __forceinline__ float red_sum16(float x) {
  x += dpp8_xor<1>(x);
  x += dpp8_xor<2>(x);
  x += dpp8_xor<4>(x);
  x += dpp16_xor8(x);
  return x;
}

__global__ __launch_bounds__(256) void sdpa_fwd_kernel(
    const float* __restrict__ Q, const float* __restrict__ K,
    const float* __restrict__ V, float* __restrict__ Out) {
  __shared__ __bf16 sK[BC * DHEAD];        // [kv][d]   row-major, 16 KB
  __shared__ __bf16 sVt[DHEAD * BC];       // [d][kv]   transposed, 16 KB
  __shared__ __bf16 sP[8][16 * BC];        // per-wave P tile, 16 KB

  const int qt   = blockIdx.x;             // q tile index (0..15)
  const int h    = blockIdx.y;             // head (0..15)
  const int tid  = threadIdx.x;
  const int w    = tid >> 5;               // wave id (0..7)
  const int lane = tid & 31;
  const int l15  = lane & 15;
  const int half = lane >> 4;              // 0: lanes 0-15, 1: lanes 16-31

  // 1/sqrt(128) * log2(e): softmax runs in exp2 domain (v_exp_f32 directly)
  const float scale = 0.08838834764831845f * 1.4426950408889634f;
  const int qwb = qt * BR + w * 16;        // this wave's first global q row

  // ---- load Q fragments (A-matrix layout, scale folded in) ----------------
  const float* qrow = Q + ((size_t)h * QLEN + (size_t)(qwb + l15)) * DHEAD;
  v16bf qa[4];
#pragma unroll
  for (int kc = 0; kc < 4; ++kc) {
    const int d0 = kc * 32 + (half << 4);
#pragma unroll
    for (int i = 0; i < 8; ++i) {
      qa[kc][i]     = f2bf(qrow[d0 + i]      * scale);
      qa[kc][i + 8] = f2bf(qrow[d0 + 8 + i] * scale);
    }
  }

  // ---- running state ------------------------------------------------------
  v8f o[8];
#pragma unroll
  for (int t = 0; t < 8; ++t) o[t] = (v8f){0.f,0.f,0.f,0.f,0.f,0.f,0.f,0.f};
  float mrow[8], lrow[8];
#pragma unroll
  for (int r = 0; r < 8; ++r) { mrow[r] = NEGBIG; lrow[r] = 0.f; }

  const float* Kg = K + (size_t)h * KVLEN * DHEAD;
  const float* Vg = V + (size_t)h * KVLEN * DHEAD;

  const int ntiles = (PREFIX + (qt + 1) * BR) / BC;   // causal trip count

  // ---- register prefetch buffers (bf16, 32 VGPRs total) -------------------
  v4bf kbuf[8], vbuf[8];

  auto prefetch = [&](int kv_base) {
#pragma unroll
    for (int i = 0; i < 8; ++i) {
      const int idx = tid + 256 * i;
      const int row = idx >> 5;            // kv within tile (32 float4 / row)
      const int c4  = (idx & 31) << 2;     // d start (multiple of 4)
      const size_t g = ((size_t)(kv_base + row) * DHEAD + c4) >> 2;
      const float4 kq = ((const float4*)Kg)[g];
      const float4 vq = ((const float4*)Vg)[g];
      kbuf[i][0] = f2bf(kq.x); kbuf[i][1] = f2bf(kq.y);
      kbuf[i][2] = f2bf(kq.z); kbuf[i][3] = f2bf(kq.w);
      vbuf[i][0] = f2bf(vq.x); vbuf[i][1] = f2bf(vq.y);
      vbuf[i][2] = f2bf(vq.z); vbuf[i][3] = f2bf(vq.w);
    }
  };
  auto commit = [&]() {
#pragma unroll
    for (int i = 0; i < 8; ++i) {
      const int idx = tid + 256 * i;
      const int row = idx >> 5;
      const int c4  = (idx & 31) << 2;
      *(v4bf*)&sK[row * DHEAD + c4] = kbuf[i];
      sVt[(c4 + 0) * BC + row] = vbuf[i][0];
      sVt[(c4 + 1) * BC + row] = vbuf[i][1];
      sVt[(c4 + 2) * BC + row] = vbuf[i][2];
      sVt[(c4 + 3) * BC + row] = vbuf[i][3];
    }
  };

  prefetch(0);

  for (int kt = 0; kt < ntiles; ++kt) {
    const int kv_base = kt * BC;

    __syncthreads();                       // previous tile fully consumed
    commit();                              // short LDS store burst
    __syncthreads();
    if (kt + 1 < ntiles) prefetch(kv_base + BC);   // overlap with compute

    // ---- S = Q K^T  (4 n-tiles x 4 d-chunks of WMMA) ---------------------
    v8f sc[4];
#pragma unroll
    for (int nt = 0; nt < 4; ++nt) {
      v8f acc = (v8f){0.f,0.f,0.f,0.f,0.f,0.f,0.f,0.f};
      const __bf16* kp = &sK[(nt * 16 + l15) * DHEAD];   // B col = kv pos
#pragma unroll
      for (int kc = 0; kc < 4; ++kc) {
        const v16bf kb = *(const v16bf*)(kp + kc * 32 + (half << 4));
        acc = __builtin_amdgcn_wmma_f32_16x16x32_bf16(
            false, qa[kc], false, kb, (short)0, acc, false, false);
      }
      sc[nt] = acc;
    }

    // ---- causal mask (diagonal tiles only) -------------------------------
    if (kv_base + BC - 1 > PREFIX + qwb) {
#pragma unroll
      for (int nt = 0; nt < 4; ++nt) {
        const int kvp = kv_base + nt * 16 + l15;
#pragma unroll
        for (int r = 0; r < 8; ++r) {
          const int qg = qwb + r + (half << 3);
          if (kvp > PREFIX + qg) sc[nt][r] = NEGBIG;
        }
      }
    }

    // ---- online softmax (exp2 domain), DPP reductions --------------------
    float alpha[8];
#pragma unroll
    for (int r = 0; r < 8; ++r) {
      float vm = fmaxf(fmaxf(sc[0][r], sc[1][r]), fmaxf(sc[2][r], sc[3][r]));
      vm = red_max16(vm);
      const float mn = fmaxf(mrow[r], vm);
      const float a  = __builtin_amdgcn_exp2f(mrow[r] - mn);
      mrow[r] = mn; alpha[r] = a;

      const int m = r + (half << 3);       // row within 16-row wave tile
      float rs = 0.f;
#pragma unroll
      for (int nt = 0; nt < 4; ++nt) {
        const float p = __builtin_amdgcn_exp2f(sc[nt][r] - mn);
        rs += p;
        sP[w][m * BC + nt * 16 + l15] = f2bf(p);
      }
      lrow[r] = lrow[r] * a + red_sum16(rs);
    }
#pragma unroll
    for (int t = 0; t < 8; ++t)
#pragma unroll
      for (int r = 0; r < 8; ++r) o[t][r] *= alpha[r];

    // P writes are wave-private; LDS is in-order per wave, just wait DScnt.
    asm volatile("s_wait_dscnt 0" ::: "memory");

    // ---- O += P V  (8 d-tiles x 2 kv-chunks of WMMA) ---------------------
    v16bf pa[2];
    const __bf16* pp = &sP[w][l15 * BC];
#pragma unroll
    for (int kc = 0; kc < 2; ++kc) {
      const int c0 = kc * 32 + (half << 4);
      const v8bf lo = *(const v8bf*)(pp + c0);
      const v8bf hi = *(const v8bf*)(pp + c0 + 8);
#pragma unroll
      for (int i = 0; i < 8; ++i) { pa[kc][i] = lo[i]; pa[kc][i + 8] = hi[i]; }
    }
#pragma unroll
    for (int t = 0; t < 8; ++t) {
      const __bf16* vp = &sVt[(t * 16 + l15) * BC];
#pragma unroll
      for (int kc = 0; kc < 2; ++kc) {
        const v16bf vb = *(const v16bf*)(vp + kc * 32 + (half << 4));
        o[t] = __builtin_amdgcn_wmma_f32_16x16x32_bf16(
            false, pa[kc], false, vb, (short)0, o[t], false, false);
      }
    }
  }

  // ---- epilogue: divide by l, write fp32 output ---------------------------
  float inv[8];
#pragma unroll
  for (int r = 0; r < 8; ++r) inv[r] = 1.0f / lrow[r];
#pragma unroll
  for (int t = 0; t < 8; ++t) {
#pragma unroll
    for (int r = 0; r < 8; ++r) {
      const int m  = r + (half << 3);
      const int qg = qwb + m;
      Out[((size_t)h * QLEN + qg) * DHEAD + t * 16 + l15] = o[t][r] * inv[r];
    }
  }
}

extern "C" void kernel_launch(void* const* d_in, const int* in_sizes, int n_in,
                              void* d_out, int out_size, void* d_ws, size_t ws_size,
                              hipStream_t stream) {
  (void)in_sizes; (void)n_in; (void)out_size; (void)d_ws; (void)ws_size;
  const float* q = (const float*)d_in[0];
  const float* k = (const float*)d_in[1];
  const float* v = (const float*)d_in[2];
  float* out = (float*)d_out;
  dim3 grid(QLEN / BR, NHEADS);
  dim3 block(256);
  sdpa_fwd_kernel<<<grid, block, 0, stream>>>(q, k, v, out);
}
